// DHGNet_49692771615012
// MI455X (gfx1250) — matrix operand order repaired and checked by reading, not compiled, not run on hardware
//
#include <hip/hip_runtime.h>
#include <math.h>

// ---- static dims (match reference) ----
#define N0_   100000
#define N1_   50000
#define NN_   150000
#define EE_   300000
#define NHID_ 256
#define HH_   16
#define DK_   16
#define RR_   4
#define TT_   2
#define LL_   2
#define DPRE_ 300
#define KPAD_ 320     // 300 zero-padded to a multiple of 32

typedef __attribute__((ext_vector_type(16))) __bf16 v16bf;
typedef __attribute__((ext_vector_type(8)))  __bf16 v8bf;
typedef __attribute__((ext_vector_type(8)))  float  v8f;

// ---------------- prep: f32 -> bf16 conversion / padding -----------------------------
__global__ __launch_bounds__(256)
void cvt_bf16_kernel(const float* __restrict__ src, __bf16* __restrict__ dst, int n8) {
    int i = blockIdx.x * blockDim.x + threadIdx.x;
    if (i >= n8) return;
    const float* s = src + (size_t)i * 8;
    float4 a = *(const float4*)s;
    float4 b = *(const float4*)(s + 4);
    v8bf o;
    o[0]=(__bf16)a.x; o[1]=(__bf16)a.y; o[2]=(__bf16)a.z; o[3]=(__bf16)a.w;
    o[4]=(__bf16)b.x; o[5]=(__bf16)b.y; o[6]=(__bf16)b.z; o[7]=(__bf16)b.w;
    *(v8bf*)(dst + (size_t)i * 8) = o;
}

// src: [nrows, 300] f32 -> dst: [nrows, 320] bf16, cols 300..319 zeroed
__global__ __launch_bounds__(256)
void pad300_bf16_kernel(const float* __restrict__ src, __bf16* __restrict__ dst, int nrows) {
    int i = blockIdx.x * blockDim.x + threadIdx.x;
    if (i >= nrows * KPAD_) return;
    int row = i / KPAD_;
    int col = i - row * KPAD_;
    float v = (col < DPRE_) ? src[(size_t)row * DPRE_ + col] : 0.0f;
    dst[i] = (__bf16)v;
}

// ---------------- WMMA tile loaders (wave32 layouts per CDNA5 ISA 7.12.2) ------------
// A 16x32 bf16 from f32 rows: lane l -> row m=l%16, K-half = l/16.
// Elements 0..7 = K kk+half*8+0..7, elements 8..15 = K kk+16+half*8+0..7.
__device__ __forceinline__ v16bf load_a_f32(const float* __restrict__ arow) {
    float4 a0 = *(const float4*)(arow);
    float4 a1 = *(const float4*)(arow + 4);
    float4 a2 = *(const float4*)(arow + 16);
    float4 a3 = *(const float4*)(arow + 20);
    v16bf A;
    A[0]=(__bf16)a0.x; A[1]=(__bf16)a0.y; A[2]=(__bf16)a0.z; A[3]=(__bf16)a0.w;
    A[4]=(__bf16)a1.x; A[5]=(__bf16)a1.y; A[6]=(__bf16)a1.z; A[7]=(__bf16)a1.w;
    A[8]=(__bf16)a2.x; A[9]=(__bf16)a2.y; A[10]=(__bf16)a2.z; A[11]=(__bf16)a2.w;
    A[12]=(__bf16)a3.x; A[13]=(__bf16)a3.y; A[14]=(__bf16)a3.z; A[15]=(__bf16)a3.w;
    return A;
}
// Same A layout but directly from a bf16 row (two aligned b128 loads, no cvt).
__device__ __forceinline__ v16bf load_a_bf16(const __bf16* __restrict__ arow) {
    v8bf lo = *(const v8bf*)(arow);
    v8bf hi = *(const v8bf*)(arow + 16);
    v16bf A;
    #pragma unroll
    for (int i = 0; i < 8; i++) { A[i] = lo[i]; A[8 + i] = hi[i]; }
    return A;
}
// B 32x16 bf16: lane l -> col n=l%16, elements 0..7 = K kk+(l/16)*8+0..7, 8..15 = K kk+16+(l/16)*8+0..7.
__device__ __forceinline__ v16bf load_b_bf16(const __bf16* __restrict__ brow) {
    v8bf lo = *(const v8bf*)(brow);
    v8bf hi = *(const v8bf*)(brow + 16);
    v16bf B;
    #pragma unroll
    for (int i = 0; i < 8; i++) { B[i] = lo[i]; B[8 + i] = hi[i]; }
    return B;
}

// ---------------- Embedding GEMM: x = pretrain @ W_trans[t]^T (bf16, K padded) ------
__global__ __launch_bounds__(32)
void embed_gemm_kernel(const __bf16* __restrict__ preb, const __bf16* __restrict__ Wtrb,
                       float* __restrict__ xout) {
    const int rb = blockIdx.x;          // 16-row block (type-pure: N0 % 16 == 0)
    const int ct = blockIdx.y;          // 64-col tile
    const int lane = threadIdx.x;
    const int m = lane & 15, half = lane >> 4;
    const int t = (rb >= (N0_/16)) ? 1 : 0;
    const __bf16* arow = preb + (size_t)(rb*16 + m) * KPAD_ + half*8;
    const __bf16* Wt   = Wtrb + (size_t)t * NHID_ * KPAD_ + (size_t)(ct*64 + m) * KPAD_ + half*8;
    v8f acc[4] = {};
    for (int kk = 0; kk < KPAD_; kk += 32) {
        v16bf A = load_a_bf16(arow + kk);
        v16bf Bt[4];
        #pragma unroll
        for (int c = 0; c < 4; c++)
            Bt[c] = load_b_bf16(Wt + (size_t)(c*16) * KPAD_ + kk);
        #pragma unroll
        for (int c = 0; c < 4; c++)
            acc[c] = __builtin_amdgcn_wmma_f32_16x16x32_bf16(false, A, false, Bt[c],
                                                             (short)0, acc[c], false, false);
    }
    #pragma unroll
    for (int c = 0; c < 4; c++) {
        const int col = ct*64 + c*16 + m;
        #pragma unroll
        for (int r = 0; r < 8; r++)
            xout[(size_t)(rb*16 + r + 8*half) * NHID_ + col] = acc[c][r];
    }
}

// ---------------- Fused per-type K/Q/V GEMM (bf16 weights, A tile reused x12) --------
__global__ __launch_bounds__(32)
void kqv_gemm_kernel(const float* __restrict__ x,
                     const __bf16* __restrict__ Wk, const __bf16* __restrict__ Wq,
                     const __bf16* __restrict__ Wv,
                     float* __restrict__ kb, float* __restrict__ qb, float* __restrict__ vb) {
    const int rb = blockIdx.x;
    const int ct = blockIdx.y;
    const int lane = threadIdx.x;
    const int m = lane & 15, half = lane >> 4;
    const int t = (rb >= (N0_/16)) ? 1 : 0;
    const size_t wbase = (size_t)t*NHID_*NHID_ + (size_t)(ct*64 + m)*NHID_ + half*8;
    const __bf16* W[3] = { Wk + wbase, Wq + wbase, Wv + wbase };
    float* outp[3] = { kb, qb, vb };
    v8f acc[3][4] = {};
    const float* arow = x + (size_t)(rb*16 + m) * NHID_ + half*8;
    for (int kk = 0; kk < NHID_; kk += 32) {
        v16bf A = load_a_f32(arow + kk);
        #pragma unroll
        for (int w = 0; w < 3; w++) {
            v16bf Bt[4];
            #pragma unroll
            for (int c = 0; c < 4; c++)
                Bt[c] = load_b_bf16(W[w] + (size_t)(c*16) * NHID_ + kk);
            #pragma unroll
            for (int c = 0; c < 4; c++)
                acc[w][c] = __builtin_amdgcn_wmma_f32_16x16x32_bf16(false, A, false, Bt[c],
                                                                    (short)0, acc[w][c],
                                                                    false, false);
        }
    }
    #pragma unroll
    for (int w = 0; w < 3; w++)
        #pragma unroll
        for (int c = 0; c < 4; c++) {
            const int col = ct*64 + c*16 + m;
            #pragma unroll
            for (int r = 0; r < 8; r++)
                outp[w][(size_t)(rb*16 + r + 8*half) * NHID_ + col] = acc[w][c][r];
        }
}

// ---------------- Single per-type GEMM (output projection Wa, bf16 weights) ----------
__global__ __launch_bounds__(32)
void type_gemm_kernel(const float* __restrict__ xin, const __bf16* __restrict__ Wa,
                      float* __restrict__ yout) {
    const int rb = blockIdx.x;
    const int ct = blockIdx.y;
    const int lane = threadIdx.x;
    const int m = lane & 15, half = lane >> 4;
    const int t = (rb >= (N0_/16)) ? 1 : 0;
    const __bf16* Wt = Wa + (size_t)t*NHID_*NHID_ + (size_t)(ct*64 + m)*NHID_ + half*8;
    v8f acc[4] = {};
    const float* arow = xin + (size_t)(rb*16 + m) * NHID_ + half*8;
    for (int kk = 0; kk < NHID_; kk += 32) {
        v16bf A = load_a_f32(arow + kk);
        v16bf Bt[4];
        #pragma unroll
        for (int c = 0; c < 4; c++)
            Bt[c] = load_b_bf16(Wt + (size_t)(c*16) * NHID_ + kk);
        #pragma unroll
        for (int c = 0; c < 4; c++)
            acc[c] = __builtin_amdgcn_wmma_f32_16x16x32_bf16(false, A, false, Bt[c],
                                                             (short)0, acc[c], false, false);
    }
    #pragma unroll
    for (int c = 0; c < 4; c++) {
        const int col = ct*64 + c*16 + m;
        #pragma unroll
        for (int r = 0; r < 8; r++)
            yout[(size_t)(rb*16 + r + 8*half) * NHID_ + col] = acc[c][r];
    }
}

// ---------------- utility / edge kernels ---------------------------------------------
__global__ void fill_kernel(float* __restrict__ p, float v, int n) {
    int i = blockIdx.x * blockDim.x + threadIdx.x;
    if (i < n) p[i] = v;
}

__device__ __forceinline__ void atomicMaxFloat(float* addr, float value) {
    if (value >= 0.0f)
        atomicMax((int*)addr, __float_as_int(value));
    else
        atomicMin((unsigned int*)addr, __float_as_uint(value));
}

// att[e,h] = (q[dst,h,:] . (k[src,h,:] @ A_att[r,h])) * pri[r,h] / sqrt(DK)
__global__ __launch_bounds__(256)
void att_kernel(const float* __restrict__ kb, const float* __restrict__ qb,
                const int* __restrict__ ei, const int* __restrict__ et,
                const float* __restrict__ Aatt, const float* __restrict__ pri,
                float* __restrict__ attb, float* __restrict__ amax) {
    int tid = blockIdx.x * blockDim.x + threadIdx.x;
    if (tid >= EE_ * HH_) return;
    int e = tid >> 4, h = tid & 15;
    int s = ei[e], d = ei[EE_ + e], r = et[e];
    const float* kv = kb + (size_t)s * NHID_ + h * DK_;
    const float* qv = qb + (size_t)d * NHID_ + h * DK_;
    const float* A  = Aatt + (size_t)(r * HH_ + h) * DK_ * DK_;   // [d][o]
    float km[DK_];
    #pragma unroll
    for (int o = 0; o < DK_; o++) km[o] = 0.0f;
    #pragma unroll 4
    for (int dd = 0; dd < DK_; dd++) {
        float kd = kv[dd];
        #pragma unroll
        for (int o = 0; o < DK_; o++) km[o] = fmaf(kd, A[dd * DK_ + o], km[o]);
    }
    float att = 0.0f;
    #pragma unroll
    for (int o = 0; o < DK_; o++) att = fmaf(qv[o], km[o], att);
    att *= pri[r * HH_ + h] * 0.25f;                               // 1/sqrt(16)
    attb[tid] = att;
    atomicMaxFloat(&amax[d * HH_ + h], att);
}

__global__ __launch_bounds__(256)
void expsum_kernel(const int* __restrict__ ei, float* __restrict__ attb,
                   const float* __restrict__ amax, float* __restrict__ denom) {
    int tid = blockIdx.x * blockDim.x + threadIdx.x;
    if (tid >= EE_ * HH_) return;
    int e = tid >> 4, h = tid & 15;
    int d = ei[EE_ + e];
    float p = __expf(attb[tid] - amax[d * HH_ + h]);
    attb[tid] = p;
    atomicAdd(&denom[d * HH_ + h], p);
}

// agg[dst,h,:] += (p/denom) * (v[src,h,:] @ A_msg[r,h])
__global__ __launch_bounds__(256)
void agg_kernel(const float* __restrict__ vb, const int* __restrict__ ei,
                const int* __restrict__ et, const float* __restrict__ Amsg,
                const float* __restrict__ attb, const float* __restrict__ denom,
                float* __restrict__ agg) {
    int tid = blockIdx.x * blockDim.x + threadIdx.x;
    if (tid >= EE_ * HH_) return;
    int e = tid >> 4, h = tid & 15;
    int s = ei[e], d = ei[EE_ + e], r = et[e];
    float w = attb[tid] / denom[d * HH_ + h];
    const float* vv = vb + (size_t)s * NHID_ + h * DK_;
    const float* A  = Amsg + (size_t)(r * HH_ + h) * DK_ * DK_;
    float vm[DK_];
    #pragma unroll
    for (int o = 0; o < DK_; o++) vm[o] = 0.0f;
    #pragma unroll 4
    for (int dd = 0; dd < DK_; dd++) {
        float vd = vv[dd];
        #pragma unroll
        for (int o = 0; o < DK_; o++) vm[o] = fmaf(vd, A[dd * DK_ + o], vm[o]);
    }
    float* ao = agg + (size_t)d * NHID_ + h * DK_;
    #pragma unroll
    for (int o = 0; o < DK_; o++) atomicAdd(&ao[o], w * vm[o]);
}

// gelu(pre) + x residual + LayerNorm, in-place on x. One wave per row, 8 feats/lane.
__global__ __launch_bounds__(256)
void post_kernel(float* __restrict__ x, const float* __restrict__ pre,
                 const float* __restrict__ gamma, const float* __restrict__ beta,
                 int nonlinear) {
    const int wid = threadIdx.x >> 5, lane = threadIdx.x & 31;
    const int row = blockIdx.x * 8 + wid;
    const float* pr = pre + (size_t)row * NHID_ + lane * 8;
    float* xr = x + (size_t)row * NHID_ + lane * 8;
    float vbuf[8];
    float4 p0 = *(const float4*)pr;
    float4 p1 = *(const float4*)(pr + 4);
    vbuf[0]=p0.x; vbuf[1]=p0.y; vbuf[2]=p0.z; vbuf[3]=p0.w;
    vbuf[4]=p1.x; vbuf[5]=p1.y; vbuf[6]=p1.z; vbuf[7]=p1.w;
    if (nonlinear) {
        #pragma unroll
        for (int i = 0; i < 8; i++) {
            float u = vbuf[i];
            vbuf[i] = 0.5f * u * (1.0f + tanhf(0.7978845608f * (u + 0.044715f * u * u * u)));
        }
    }
    float4 x0 = *(const float4*)xr;
    float4 x1 = *(const float4*)(xr + 4);
    vbuf[0]+=x0.x; vbuf[1]+=x0.y; vbuf[2]+=x0.z; vbuf[3]+=x0.w;
    vbuf[4]+=x1.x; vbuf[5]+=x1.y; vbuf[6]+=x1.z; vbuf[7]+=x1.w;
    float s = 0.0f, ss = 0.0f;
    #pragma unroll
    for (int i = 0; i < 8; i++) { s += vbuf[i]; ss += vbuf[i] * vbuf[i]; }
    #pragma unroll
    for (int off = 16; off >= 1; off >>= 1) {
        s  += __shfl_xor(s,  off, 32);
        ss += __shfl_xor(ss, off, 32);
    }
    float mean = s * (1.0f / NHID_);
    float var  = ss * (1.0f / NHID_) - mean * mean;
    float rstd = rsqrtf(var + 1e-5f);
    float o[8];
    #pragma unroll
    for (int i = 0; i < 8; i++)
        o[i] = (vbuf[i] - mean) * rstd * gamma[lane*8 + i] + beta[lane*8 + i];
    *(float4*)xr       = make_float4(o[0], o[1], o[2], o[3]);
    *(float4*)(xr + 4) = make_float4(o[4], o[5], o[6], o[7]);
}

// out[b,:] = (word_idx[b] == PAD) ? 0 : x[word_idx[b], :]
__global__ __launch_bounds__(64)
void gather_kernel(const int* __restrict__ widx, const float* __restrict__ x,
                   float* __restrict__ out) {
    const int b = blockIdx.x;
    const int idx = widx[b];
    float4* o = (float4*)(out + (size_t)b * NHID_);
    if (idx == 0) {
        o[threadIdx.x] = make_float4(0.f, 0.f, 0.f, 0.f);
    } else {
        const float4* srcr = (const float4*)(x + (size_t)idx * NHID_);
        o[threadIdx.x] = srcr[threadIdx.x];
    }
}

// ---------------- host-side orchestration --------------------------------------------
extern "C" void kernel_launch(void* const* d_in, const int* in_sizes, int n_in,
                              void* d_out, int out_size, void* d_ws, size_t ws_size,
                              hipStream_t stream) {
    (void)in_sizes; (void)n_in; (void)out_size; (void)ws_size;
    const int*   word_idx   = (const int*)d_in[0];
    const int*   edge_index = (const int*)d_in[1];
    const int*   edge_type  = (const int*)d_in[2];
    // d_in[3] = node_type: unused (types are contiguous blocks, N0 % 16 == 0)
    const float* pre0    = (const float*)d_in[4];
    const float* pre1    = (const float*)d_in[5];
    const float* Wtr     = (const float*)d_in[6];
    const float* Wk      = (const float*)d_in[7];
    const float* Wq      = (const float*)d_in[8];
    const float* Wv      = (const float*)d_in[9];
    const float* Wa      = (const float*)d_in[10];
    const float* rel_att = (const float*)d_in[11];
    const float* rel_msg = (const float*)d_in[12];
    const float* rel_pri = (const float*)d_in[13];
    const float* ln_g    = (const float*)d_in[14];
    const float* ln_b    = (const float*)d_in[15];
    float* out = (float*)d_out;

    // f32 scratch
    float* ws    = (float*)d_ws;
    float* x     = ws;                                 // [N,256]
    float* kb    = x    + (size_t)NN_ * NHID_;         // [N,256] (also reused as `pre`)
    float* qb    = kb   + (size_t)NN_ * NHID_;
    float* vb    = qb   + (size_t)NN_ * NHID_;
    float* agg   = vb   + (size_t)NN_ * NHID_;
    float* attb  = agg  + (size_t)NN_ * NHID_;         // [E,16]
    float* amax  = attb + (size_t)EE_ * HH_;           // [N,16]
    float* denom = amax + (size_t)NN_ * HH_;           // [N,16]
    // bf16 scratch (16/32-byte aligned: f32 region is 806,400,000 B, /32 exact)
    __bf16* preb = (__bf16*)(denom + (size_t)NN_ * HH_);   // [N,320] padded pretrain
    __bf16* Wtrb = preb + (size_t)NN_ * KPAD_;             // [T,256,320]
    __bf16* Wkb  = Wtrb + (size_t)TT_ * NHID_ * KPAD_;     // [L,T,256,256]
    __bf16* Wqb  = Wkb  + (size_t)LL_ * TT_ * NHID_ * NHID_;
    __bf16* Wvb  = Wqb  + (size_t)LL_ * TT_ * NHID_ * NHID_;
    __bf16* Wab  = Wvb  + (size_t)LL_ * TT_ * NHID_ * NHID_;

    // --- prep: pad/convert activations + weights to bf16 ---
    {
        int n0 = N0_ * KPAD_, n1 = N1_ * KPAD_, ntr = TT_ * NHID_ * KPAD_;
        pad300_bf16_kernel<<<(n0 + 255)/256, 256, 0, stream>>>(pre0, preb, N0_);
        pad300_bf16_kernel<<<(n1 + 255)/256, 256, 0, stream>>>(pre1, preb + (size_t)N0_*KPAD_, N1_);
        pad300_bf16_kernel<<<(ntr + 255)/256, 256, 0, stream>>>(Wtr, Wtrb, TT_ * NHID_);
        int nw8 = (LL_ * TT_ * NHID_ * NHID_) / 8;
        cvt_bf16_kernel<<<(nw8 + 255)/256, 256, 0, stream>>>(Wk, Wkb, nw8);
        cvt_bf16_kernel<<<(nw8 + 255)/256, 256, 0, stream>>>(Wq, Wqb, nw8);
        cvt_bf16_kernel<<<(nw8 + 255)/256, 256, 0, stream>>>(Wv, Wvb, nw8);
        cvt_bf16_kernel<<<(nw8 + 255)/256, 256, 0, stream>>>(Wa, Wab, nw8);
    }

    const dim3 gemm_grid(NN_ / 16, 4);
    embed_gemm_kernel<<<gemm_grid, 32, 0, stream>>>(preb, Wtrb, x);

    const int EH = EE_ * HH_;
    const int NH = NN_ * HH_;
    for (int l = 0; l < LL_; l++) {
        const size_t woff = (size_t)l * TT_ * NHID_ * NHID_;
        const float* Aatt = rel_att + (size_t)l * RR_ * HH_ * DK_ * DK_;
        const float* Amsg = rel_msg + (size_t)l * RR_ * HH_ * DK_ * DK_;
        const float* pri  = rel_pri + (size_t)l * RR_ * HH_;

        kqv_gemm_kernel<<<gemm_grid, 32, 0, stream>>>(x, Wkb + woff, Wqb + woff, Wvb + woff,
                                                      kb, qb, vb);
        fill_kernel<<<(NH + 255) / 256, 256, 0, stream>>>(amax, -1e30f, NH);
        fill_kernel<<<(NH + 255) / 256, 256, 0, stream>>>(denom, 0.0f, NH);
        fill_kernel<<<(NN_ * NHID_ + 255) / 256, 256, 0, stream>>>(agg, 0.0f, NN_ * NHID_);
        att_kernel<<<(EH + 255) / 256, 256, 0, stream>>>(kb, qb, edge_index, edge_type,
                                                         Aatt, pri, attb, amax);
        expsum_kernel<<<(EH + 255) / 256, 256, 0, stream>>>(edge_index, attb, amax, denom);
        agg_kernel<<<(EH + 255) / 256, 256, 0, stream>>>(vb, edge_index, edge_type,
                                                         Amsg, attb, denom, agg);
        type_gemm_kernel<<<gemm_grid, 32, 0, stream>>>(agg, Wab + woff, kb);   // pre := kb
        post_kernel<<<NN_ / 8, 256, 0, stream>>>(x, kb, ln_g + l * NHID_, ln_b + l * NHID_,
                                                 (l < LL_ - 1) ? 1 : 0);
    }
    gather_kernel<<<32 * 1024, 64, 0, stream>>>(word_idx, x, out);
}
